// EdgeNoiseGenerator_87686052315842
// MI455X (gfx1250) — matrix operand order repaired and checked by reading, not compiled, not run on hardware
//
#include <hip/hip_runtime.h>
#include <hip/hip_bf16.h>

typedef __attribute__((ext_vector_type(16))) _Float16 v16h;
typedef __attribute__((ext_vector_type(8)))  float    v8f;
typedef __attribute__((ext_vector_type(4)))  _Float16 v4h;

#define D_IN 64

// ---------------------------------------------------------------------------
// Kernel 0: convert weights fp32->f16 and pack into per-lane B-operand layout.
// One B operand (16x16x32 WMMA: B is 32x16) = 32 lanes * 16 halves = 512 halves,
// stored lane-contiguous so a wave reads it as one coalesced v16h per lane.
// Element: B[k][n] with k_local = 16*(lane>>4) + h, n = 16*nt + (lane&15).
// ---------------------------------------------------------------------------
__global__ void pack_weights_kernel(const float* __restrict__ W1,   // [64][128]
                                    const float* __restrict__ W2,   // [128][64]
                                    const float* __restrict__ We1,  // [128][128]
                                    _Float16* __restrict__ ws) {
  _Float16* w1p = ws;            // 2 kb * 8 nt  = 16 ops * 512
  _Float16* w2p = ws + 8192;     // 4 kb * 4 nt  = 16 ops * 512
  _Float16* wtp = ws + 16384;    // We1 top: 2 kb * 8 nt
  _Float16* wbp = ws + 24576;    // We1 bot: 2 kb * 8 nt
  const int tid = threadIdx.x;

  // W1: K=64 (kb 0..1), N=128 (nt 0..7), ld=128
  for (int idx = tid; idx < 16 * 512; idx += blockDim.x) {
    int p = idx >> 9, r = idx & 511, l = r >> 4, h = r & 15;
    int kb = p >> 3, nt = p & 7;
    int k = kb * 32 + ((l >> 4) << 4) + h;
    int n = nt * 16 + (l & 15);
    w1p[idx] = (_Float16)W1[k * 128 + n];
  }
  // W2: K=128 (kb 0..3), N=64 (nt 0..3), ld=64
  for (int idx = tid; idx < 16 * 512; idx += blockDim.x) {
    int p = idx >> 9, r = idx & 511, l = r >> 4, h = r & 15;
    int kb = p >> 2, nt = p & 3;
    int k = kb * 32 + ((l >> 4) << 4) + h;
    int n = nt * 16 + (l & 15);
    w2p[idx] = (_Float16)W2[k * 64 + n];
  }
  // We1 split: top rows 0..63, bottom rows 64..127; N=128 (nt 0..7), ld=128
  for (int idx = tid; idx < 16 * 512; idx += blockDim.x) {
    int p = idx >> 9, r = idx & 511, l = r >> 4, h = r & 15;
    int kb = p >> 3, nt = p & 7;
    int k = kb * 32 + ((l >> 4) << 4) + h;
    int n = nt * 16 + (l & 15);
    wtp[idx] = (_Float16)We1[k * 128 + n];
    wbp[idx] = (_Float16)We1[(64 + k) * 128 + n];
  }
}

// ---------------------------------------------------------------------------
// Kernel 1: node pipeline, one wave per 16-node tile, all GEMMs via WMMA.
//   H1   = relu(x @ W1 + b1)            16x128
//   feat = H1 @ W2 + b2                 16x64
//   Psrc = feat @ We1[:64]  + be1       16x128  (f16 table)
//   Pdst = feat @ We1[64:]              16x128  (f16 table)
// ---------------------------------------------------------------------------
__global__ __launch_bounds__(256) void node_kernel(
    const float* __restrict__ x, const float* __restrict__ b1,
    const float* __restrict__ b2, const float* __restrict__ be1,
    const _Float16* __restrict__ ws, _Float16* __restrict__ Psrc,
    _Float16* __restrict__ Pdst, int ntiles) {
  __shared__ _Float16 h1s[8][16 * 128];  // 32 KB: per-wave H1 staging
  __shared__ _Float16 fts[8][16 * 64];   // 16 KB: per-wave feat staging

  const int lane = threadIdx.x & 31;
  const int wid  = threadIdx.x >> 5;
  const int tile = blockIdx.x * 8 + wid;
  if (tile >= ntiles) return;            // uniform per wave -> EXEC stays all-1

  const int m   = lane & 15;             // row (A/C) or col (B/C) index
  const int hi8 = (lane >> 4) << 3;      // +8 for upper half-wave (C row; A K-offset is hi8<<1 = 16)
  const int node0 = tile * 16;

  const _Float16* w1p = ws;
  const _Float16* w2p = ws + 8192;
  const _Float16* wtp = ws + 16384;
  const _Float16* wbp = ws + 24576;

  // ---- Layer 1: A = x tile (K=64, 2 K-blocks), packed straight from global
  v16h A1[2];
#pragma unroll
  for (int kb = 0; kb < 2; ++kb) {
#pragma unroll
    for (int h = 0; h < 16; ++h) {
      int k = kb * 32 + (hi8 << 1) + h;
      A1[kb][h] = (_Float16)x[(node0 + m) * D_IN + k];
    }
  }
#pragma unroll
  for (int nt = 0; nt < 8; ++nt) {
    v8f c = {};
#pragma unroll
    for (int kb = 0; kb < 2; ++kb) {
      v16h b = *(const v16h*)(w1p + (kb * 8 + nt) * 512 + lane * 16);
      c = __builtin_amdgcn_wmma_f32_16x16x32_f16(false, A1[kb], false, b,
                                                 (short)0, c, false, false);
    }
    float bias = b1[nt * 16 + m];
#pragma unroll
    for (int r = 0; r < 8; ++r) {
      float v = c[r] + bias;
      v = v > 0.0f ? v : 0.0f;
      h1s[wid][(r + hi8) * 128 + nt * 16 + m] = (_Float16)v;
    }
  }

  // ---- Layer 2: A = H1 (K=128, 4 K-blocks) from LDS staging
  v16h A2[4];
#pragma unroll
  for (int kb = 0; kb < 4; ++kb) {
#pragma unroll
    for (int h = 0; h < 16; ++h) {
      int k = kb * 32 + (hi8 << 1) + h;
      A2[kb][h] = h1s[wid][m * 128 + k];
    }
  }
#pragma unroll
  for (int nt = 0; nt < 4; ++nt) {
    v8f c = {};
#pragma unroll
    for (int kb = 0; kb < 4; ++kb) {
      v16h b = *(const v16h*)(w2p + (kb * 4 + nt) * 512 + lane * 16);
      c = __builtin_amdgcn_wmma_f32_16x16x32_f16(false, A2[kb], false, b,
                                                 (short)0, c, false, false);
    }
    float bias = b2[nt * 16 + m];
#pragma unroll
    for (int r = 0; r < 8; ++r) {
      fts[wid][(r + hi8) * 64 + nt * 16 + m] = (_Float16)(c[r] + bias);
    }
  }

  // ---- Layer 3: A = feat (K=64, 2 K-blocks); two projections (src/dst halves)
  v16h A3[2];
#pragma unroll
  for (int kb = 0; kb < 2; ++kb) {
#pragma unroll
    for (int h = 0; h < 16; ++h) {
      int k = kb * 32 + (hi8 << 1) + h;
      A3[kb][h] = fts[wid][m * 64 + k];
    }
  }
#pragma unroll
  for (int nt = 0; nt < 8; ++nt) {
    v8f cs = {};
    v8f cd = {};
#pragma unroll
    for (int kb = 0; kb < 2; ++kb) {
      v16h bt = *(const v16h*)(wtp + (kb * 8 + nt) * 512 + lane * 16);
      cs = __builtin_amdgcn_wmma_f32_16x16x32_f16(false, A3[kb], false, bt,
                                                  (short)0, cs, false, false);
      v16h bb = *(const v16h*)(wbp + (kb * 8 + nt) * 512 + lane * 16);
      cd = __builtin_amdgcn_wmma_f32_16x16x32_f16(false, A3[kb], false, bb,
                                                  (short)0, cd, false, false);
    }
    float bias = be1[nt * 16 + m];  // fold be1 into Psrc only (appears once)
#pragma unroll
    for (int r = 0; r < 8; ++r) {
      size_t row = (size_t)(node0 + r + hi8) * 128 + nt * 16 + m;
      Psrc[row] = (_Float16)(cs[r] + bias);
      Pdst[row] = (_Float16)(cd[r]);
    }
  }
}

// ---------------------------------------------------------------------------
// Kernel 2: per-edge: out[e] = relu(Psrc[src] + Pdst[dst]) . We2 + be2
// One wave per edge; lane L covers features 4L..4L+3 (8B coalesced loads,
// whole 256B f16 row per gather), then 5-step shfl_xor wave reduction.
// Tables (2 x 25.6 MB) are L2-resident on MI455X (192 MB L2).
// ---------------------------------------------------------------------------
__global__ __launch_bounds__(256) void edge_kernel(
    const int* __restrict__ edge_index, const _Float16* __restrict__ Psrc,
    const _Float16* __restrict__ Pdst, const float* __restrict__ We2,
    const float* __restrict__ be2, float* __restrict__ out, int E) {
  const int lane = threadIdx.x & 31;
  const int wave = (blockIdx.x * blockDim.x + threadIdx.x) >> 5;
  const int nwaves = (gridDim.x * blockDim.x) >> 5;

  const float4 w = ((const float4*)We2)[lane];  // We2[4*lane .. 4*lane+3]
  const float bout = be2[0];
  const int* __restrict__ srcI = edge_index;
  const int* __restrict__ dstI = edge_index + E;

  for (int e = wave; e < E; e += nwaves) {
    const int s = srcI[e];
    const int d = dstI[e];
    v4h ps = *(const v4h*)(Psrc + (size_t)s * 128 + lane * 4);
    v4h pd = *(const v4h*)(Pdst + (size_t)d * 128 + lane * 4);
    float h0 = (float)ps[0] + (float)pd[0]; h0 = h0 > 0.f ? h0 : 0.f;
    float h1 = (float)ps[1] + (float)pd[1]; h1 = h1 > 0.f ? h1 : 0.f;
    float h2 = (float)ps[2] + (float)pd[2]; h2 = h2 > 0.f ? h2 : 0.f;
    float h3 = (float)ps[3] + (float)pd[3]; h3 = h3 > 0.f ? h3 : 0.f;
    float acc = h0 * w.x + h1 * w.y + h2 * w.z + h3 * w.w;
#pragma unroll
    for (int off = 16; off > 0; off >>= 1) acc += __shfl_xor(acc, off, 32);
    if (lane == 0) out[e] = acc + bout;
  }
}

// ---------------------------------------------------------------------------
extern "C" void kernel_launch(void* const* d_in, const int* in_sizes, int n_in,
                              void* d_out, int out_size, void* d_ws,
                              size_t ws_size, hipStream_t stream) {
  const float* x   = (const float*)d_in[0];
  const int*   ei  = (const int*)d_in[1];
  const float* W1  = (const float*)d_in[2];
  const float* b1  = (const float*)d_in[3];
  const float* W2  = (const float*)d_in[4];
  const float* b2  = (const float*)d_in[5];
  const float* We1 = (const float*)d_in[6];
  const float* be1 = (const float*)d_in[7];
  const float* We2 = (const float*)d_in[8];
  const float* be2 = (const float*)d_in[9];
  float* out = (float*)d_out;

  const int N = in_sizes[0] / D_IN;   // 100000 nodes
  const int E = in_sizes[1] / 2;      // 1000000 edges

  _Float16* ws   = (_Float16*)d_ws;
  _Float16* Psrc = ws + 32768;                    // after 64 KB of packed weights
  _Float16* Pdst = Psrc + (size_t)N * 128;

  pack_weights_kernel<<<1, 256, 0, stream>>>(W1, W2, We1, ws);

  const int ntiles = (N + 15) / 16;               // 6250 (exact)
  const int blocksA = (ntiles + 7) / 8;
  node_kernel<<<blocksA, 256, 0, stream>>>(x, b1, b2, be1, ws, Psrc, Pdst,
                                           ntiles);

  edge_kernel<<<4096, 256, 0, stream>>>(ei, Psrc, Pdst, We2, be2, out, E);
}
